// MultiHeadAttention_86878598464357
// MI455X (gfx1250) — matrix operand order repaired and checked by reading, not compiled, not run on hardware
//
#include <hip/hip_runtime.h>
#include <hip/hip_bf16.h>

typedef __attribute__((ext_vector_type(16))) _Float16 v16h;
typedef __attribute__((ext_vector_type(8)))  _Float16 v8h;
typedef __attribute__((ext_vector_type(8)))  float    v8f;

#define DEVI static __device__ __forceinline__

// ---- Tensor Data Mover availability (device pass only) ----------------------
#if defined(__HIP_DEVICE_COMPILE__)
#if __has_builtin(__builtin_amdgcn_tensor_load_to_lds) && \
    __has_builtin(__builtin_amdgcn_s_wait_tensorcnt)
#define USE_TDM 1
#endif
#endif

#ifdef USE_TDM
typedef unsigned int u32x4 __attribute__((ext_vector_type(4)));
typedef int          i32x4 __attribute__((ext_vector_type(4)));
typedef int          i32x8 __attribute__((ext_vector_type(8)));

// 2-D f16 tile DMA: global (row-major, row_stride elems) -> LDS, with LDS row
// padding produced by the TDM itself (pad_interval/pad_amount codes per D#).
DEVI void tdm_load_tile_f16(unsigned lds_byte_addr, const _Float16* gptr,
                            unsigned tile_w, unsigned tile_h,
                            unsigned tensor_w, unsigned tensor_h,
                            unsigned row_stride,
                            unsigned pad_interval_code, unsigned pad_amount_code)
{
  const unsigned long long ga = (unsigned long long)(uintptr_t)gptr;
  u32x4 g0;
  g0[0] = 1u;                                            // count=1, user D#
  g0[1] = lds_byte_addr;                                 // lds_addr
  g0[2] = (unsigned)(ga & 0xffffffffull);                // global_addr[31:0]
  g0[3] = (unsigned)((ga >> 32) & 0x1ffffffull)          // global_addr[56:32]
        | (2u << 30);                                    // type = 2 ("image")
  i32x8 g1;
  g1[0] = (int)((1u << 16)                               // data_size = 2 bytes
        | (1u << 20)                                     // pad_enable
        | (pad_interval_code << 22)
        | (pad_amount_code << 25));
  g1[1] = (int)((tensor_w & 0xffffu) << 16);             // tensor_dim0[15:0]
  g1[2] = (int)(((tensor_w >> 16) & 0xffffu)             // tensor_dim0[31:16]
        | ((tensor_h & 0xffffu) << 16));                 // tensor_dim1[15:0]
  g1[3] = (int)(((tensor_h >> 16) & 0xffffu)             // tensor_dim1[31:16]
        | ((tile_w & 0xffffu) << 16));                   // tile_dim0
  g1[4] = (int)(tile_h & 0xffffu);                       // tile_dim1 (dim2=0)
  g1[5] = (int)row_stride;                               // tensor_dim0_stride lo
  g1[6] = 0;                                             // stride hi / d1stride lo
  g1[7] = 0;
  i32x4 z4 = (i32x4)0;
#if __clang_major__ >= 23
  i32x8 z8 = (i32x8)0;
  __builtin_amdgcn_tensor_load_to_lds(g0, g1, z4, z4, z8, 0);
#else
  __builtin_amdgcn_tensor_load_to_lds(g0, g1, z4, z4, 0);
#endif
}
#endif  // USE_TDM

// D = A(16x32 f16) * B(32x16 f16) + C(16x16 f32)
DEVI v8f wmma_f16(v16h a, v16h b, v8f c) {
  return __builtin_amdgcn_wmma_f32_16x16x32_f16(
      false, a, false, b, (short)0, c, false, false);
}

DEVI v16h cat8(v8h lo, v8h hi) {
  return __builtin_shufflevector(lo, hi, 0,1,2,3,4,5,6,7,8,9,10,11,12,13,14,15);
}

// A fragment (16x32 f16) from LDS, row stride ld halves.
// ISA 7.12.2: lane<16 -> K {0..7,16..23}; lane>=16 -> K {8..15,24..31}; M=lane%16.
DEVI v16h ldsA_frag(const _Float16* sA, int ld, int row0, int lane) {
  const int m  = row0 + (lane & 15);
  const int ko = (lane < 16) ? 0 : 8;
  const _Float16* p = sA + m * ld + ko;
  v8h lo = *(const v8h*)(p);
  v8h hi = *(const v8h*)(p + 16);
  return cat8(lo, hi);
}

// B fragment (32x16 f16) from LDS laid out [K][N]: N = lane%16, K in table order.
DEVI v16h ldsB_frag(const _Float16* sB, int ld, int n0, int lane) {
  const int n = n0 + (lane & 15), ko = (lane < 16) ? 0 : 8;
  v16h r;
  #pragma unroll
  for (int s = 0; s < 16; s++) r[s] = sB[(ko + (s < 8 ? s : s + 8)) * ld + n];
  return r;
}

// A fragment from row-major global f16.
DEVI v16h gA_frag(const _Float16* base, int ld, int row0, int k0, int lane) {
  const _Float16* p = base + (size_t)(row0 + (lane & 15)) * ld + k0 + ((lane < 16) ? 0 : 8);
  v8h lo = *(const v8h*)(p);
  v8h hi = *(const v8h*)(p + 16);
  return cat8(lo, hi);
}

// ---------------------------------------------------------------------------
// Kernel 1: qkv = x @ w_qkv^T with fused RoPE.
// 128(token) x 128(feature) tile == one head of one of {q,k,v}.
// Software-pipelined: next k-tile fetched to registers during WMMA phase.
// ---------------------------------------------------------------------------
__global__ __launch_bounds__(256) void qkv_rope_kernel(
    const float* __restrict__ x, const float* __restrict__ wqkv,
    _Float16* __restrict__ Qh, _Float16* __restrict__ KTh, _Float16* __restrict__ Vh)
{
  __shared__ _Float16 sA[128 * 40];    // x tile   [128][32] pad->40
  __shared__ _Float16 sB[32 * 136];    // w tile^T [32][128] pad->136
  __shared__ _Float16 sC[128 * 132];   // staging  [128][128] pad->132

  const int tid  = threadIdx.x;
  const int lane = tid & 31;
  const int wv   = tid >> 5;
  const int m0   = blockIdx.y * 128;
  const int f0   = blockIdx.x * 128;

  const int mBase = (wv >> 1) * 32;
  const int nBase = (wv & 1) * 64;

  v8f acc[2][4] = {};
  float4 rA[4], rB[4];

  // prologue: fetch k-tile 0
  #pragma unroll
  for (int u = 0; u < 4; u++) {
    const int cc = tid + 256 * u;
    rA[u] = *(const float4*)(x    + (size_t)(m0 + (cc >> 3)) * 2048 + (cc & 7) * 4);
    rB[u] = *(const float4*)(wqkv + (size_t)(f0 + (cc >> 3)) * 2048 + (cc & 7) * 4);
  }

  for (int k0 = 0; k0 < 2048; k0 += 32) {
    __syncthreads();
    // regs -> LDS (A: f16 rows; B: f16 transposed [k][n])
    #pragma unroll
    for (int u = 0; u < 4; u++) {
      const int cc = tid + 256 * u;
      const int r = cc >> 3, c4 = (cc & 7) * 4;
      _Float16* d = sA + r * 40 + c4;
      d[0] = (_Float16)rA[u].x; d[1] = (_Float16)rA[u].y;
      d[2] = (_Float16)rA[u].z; d[3] = (_Float16)rA[u].w;
      sB[(c4 + 0) * 136 + r] = (_Float16)rB[u].x;
      sB[(c4 + 1) * 136 + r] = (_Float16)rB[u].y;
      sB[(c4 + 2) * 136 + r] = (_Float16)rB[u].z;
      sB[(c4 + 3) * 136 + r] = (_Float16)rB[u].w;
    }
    __syncthreads();
    // overlap: fetch next k-tile while WMMAs run on this one
    if (k0 + 32 < 2048) {
      #pragma unroll
      for (int u = 0; u < 4; u++) {
        const int cc = tid + 256 * u;
        rA[u] = *(const float4*)(x    + (size_t)(m0 + (cc >> 3)) * 2048 + k0 + 32 + (cc & 7) * 4);
        rB[u] = *(const float4*)(wqkv + (size_t)(f0 + (cc >> 3)) * 2048 + k0 + 32 + (cc & 7) * 4);
      }
    }
    v16h af[2], bf[4];
    #pragma unroll
    for (int i = 0; i < 2; i++) af[i] = ldsA_frag(sA, 40, mBase + 16 * i, lane);
    #pragma unroll
    for (int j = 0; j < 4; j++) bf[j] = ldsB_frag(sB, 136, nBase + 16 * j, lane);
    #pragma unroll
    for (int i = 0; i < 2; i++)
      #pragma unroll
      for (int j = 0; j < 4; j++)
        acc[i][j] = wmma_f16(af[i], bf[j], acc[i][j]);
  }

  __syncthreads();
  const int cl = lane & 15, hi8 = (lane >> 4) * 8;
  #pragma unroll
  for (int i = 0; i < 2; i++)
    #pragma unroll
    for (int j = 0; j < 4; j++)
      #pragma unroll
      for (int r = 0; r < 8; r++)
        sC[(mBase + 16 * i + r + hi8) * 132 + nBase + 16 * j + cl] =
            (_Float16)acc[i][j][r];
  __syncthreads();

  const int which = f0 >> 11;          // 0=q 1=k 2=v
  const int h     = (f0 & 2047) >> 7;
  const float qscale = 0.08838834764831845f;   // 1/sqrt(128)

  if (which == 2) {
    for (int idx = tid; idx < 128 * 128; idx += 256) {
      const int r = idx >> 7, d = idx & 127;
      const int bt = m0 + r, b = bt >> 11, t = bt & 2047;
      Vh[(((size_t)(b * 16 + h)) * 2048 + t) * 128 + d] = sC[r * 132 + d];
    }
  } else {
    for (int idx = tid; idx < 128 * 64; idx += 256) {
      const int r = idx >> 6, d = idx & 63;
      const int bt = m0 + r, b = bt >> 11, t = bt & 2047;
      const float t1 = (float)sC[r * 132 + d];
      const float t2 = (float)sC[r * 132 + d + 64];
      const float freq = __powf(10000.0f, -(float)d * (1.0f / 64.0f));
      float s, c;
      __sincosf((float)t * freq, &s, &c);
      const float o1 = t1 * c - t2 * s;
      const float o2 = t2 * c + t1 * s;
      const int bh = b * 16 + h;
      if (which == 0) {
        const size_t o = ((size_t)bh * 2048 + t) * 128 + d;
        Qh[o]      = (_Float16)(o1 * qscale);
        Qh[o + 64] = (_Float16)(o2 * qscale);
      } else {
        const size_t o = ((size_t)bh * 128 + d) * 2048 + t;  // K^T [D][T]
        KTh[o]             = (_Float16)o1;
        KTh[o + 64 * 2048] = (_Float16)o2;
      }
    }
  }
}

// ---------------------------------------------------------------------------
// Kernel 2: causal flash attention per (b,h, 64-row Q tile).
// Softmax parallelized 4 threads/row with wave32 shfl_xor reductions.
// ---------------------------------------------------------------------------
__global__ __launch_bounds__(256) void flash_attn_kernel(
    const _Float16* __restrict__ Qh, const _Float16* __restrict__ KTh,
    const _Float16* __restrict__ Vh, _Float16* __restrict__ AOh)
{
  __shared__ float    sS[64 * 68];
  __shared__ _Float16 sP[64 * 72];
  __shared__ float sM[64], sL[64], sAlpha[64];

  const int tid  = threadIdx.x;
  const int lane = tid & 31;
  const int wv   = tid >> 5;
  const int i0   = blockIdx.x * 64;
  const int bh   = blockIdx.y;

  const _Float16* Qp  = Qh  + (size_t)bh * 2048 * 128;  // [T][D]
  const _Float16* KTp = KTh + (size_t)bh * 128 * 2048;  // [D][T]
  const _Float16* Vp  = Vh  + (size_t)bh * 2048 * 128;  // [T][D]

  const int smB = (wv >> 2) * 32;
  const int snB = (wv & 3) * 16;
  const int odB = (wv & 3) * 32;

  v16h qf[2][4];
  #pragma unroll
  for (int i = 0; i < 2; i++)
    #pragma unroll
    for (int ks = 0; ks < 4; ks++)
      qf[i][ks] = gA_frag(Qp, 128, i0 + smB + 16 * i, ks * 32, lane);

  v8f oacc[2][2] = {};
  if (tid < 64) { sM[tid] = -1e30f; sL[tid] = 0.0f; }

  const int cl = lane & 15, hi8 = (lane >> 4) * 8;

  for (int j0 = 0; j0 <= i0; j0 += 64) {
    // hint the next key tile toward the caches while we compute this one
    if (j0 + 64 <= i0) {
      __builtin_prefetch(KTp + (size_t)lane * 2048 + j0 + 64 + snB, 0, 1);
      __builtin_prefetch(Vp + (size_t)(j0 + 64 + lane) * 128 + odB, 0, 1);
    }
    // ---- S = Q * K^T ----
    v8f sacc[2] = {};
    #pragma unroll
    for (int ks = 0; ks < 4; ks++) {
      v16h bf;
      #pragma unroll
      for (int s = 0; s < 16; s++)
        bf[s] = KTp[(size_t)(ks * 32 + ((lane < 16) ? 0 : 8) + (s < 8 ? s : s + 8)) * 2048 + j0 + snB + cl];
      #pragma unroll
      for (int i = 0; i < 2; i++)
        sacc[i] = wmma_f16(qf[i][ks], bf, sacc[i]);
    }
    #pragma unroll
    for (int i = 0; i < 2; i++)
      #pragma unroll
      for (int r = 0; r < 8; r++)
        sS[(smB + 16 * i + r + hi8) * 68 + snB + cl] = sacc[i][r];
    __syncthreads();

    // ---- online softmax: 4 threads per row, 16 cols each ----
    {
      const int row = tid >> 2;
      const int c0  = (tid & 3) * 16;
      const int qi  = i0 + row;
      const float mOld = sM[row];
      float rm = -1e30f;
      #pragma unroll
      for (int c = 0; c < 16; c++)
        if (j0 + c0 + c <= qi) rm = fmaxf(rm, sS[row * 68 + c0 + c]);
      rm = fmaxf(rm, __shfl_xor(rm, 1));
      rm = fmaxf(rm, __shfl_xor(rm, 2));
      const float mNew  = fmaxf(mOld, rm);
      const float alpha = __expf(mOld - mNew);
      float lsum = 0.0f;
      #pragma unroll
      for (int c = 0; c < 16; c++) {
        const float p = (j0 + c0 + c <= qi)
                          ? __expf(sS[row * 68 + c0 + c] - mNew) : 0.0f;
        sP[row * 72 + c0 + c] = (_Float16)p;
        lsum += p;
      }
      lsum += __shfl_xor(lsum, 1);
      lsum += __shfl_xor(lsum, 2);
      if ((tid & 3) == 0) {
        sM[row]     = mNew;
        sL[row]     = sL[row] * alpha + lsum;
        sAlpha[row] = alpha;
      }
    }
    __syncthreads();

    // ---- rescale O, accumulate O += P * V ----
    #pragma unroll
    for (int i = 0; i < 2; i++)
      #pragma unroll
      for (int r = 0; r < 8; r++) {
        const float a = sAlpha[smB + 16 * i + r + hi8];
        oacc[i][0][r] *= a;
        oacc[i][1][r] *= a;
      }
    #pragma unroll
    for (int ks = 0; ks < 2; ks++) {
      v16h vf[2];
      #pragma unroll
      for (int j = 0; j < 2; j++)
        #pragma unroll
        for (int s = 0; s < 16; s++)
          vf[j][s] = Vp[(size_t)(j0 + ks * 32 + ((lane < 16) ? 0 : 8) + (s < 8 ? s : s + 8)) * 128 + odB + 16 * j + cl];
      v16h pf[2];
      #pragma unroll
      for (int i = 0; i < 2; i++) {
        const int m  = smB + 16 * i + cl;
        const int ko = ks * 32 + ((lane < 16) ? 0 : 8);
        v8h lo = *(const v8h*)(sP + m * 72 + ko);
        v8h hi = *(const v8h*)(sP + m * 72 + ko + 16);
        pf[i] = cat8(lo, hi);
      }
      #pragma unroll
      for (int i = 0; i < 2; i++)
        #pragma unroll
        for (int j = 0; j < 2; j++)
          oacc[i][j] = wmma_f16(pf[i], vf[j], oacc[i][j]);
    }
  }

  const int b = bh >> 4, h = bh & 15;
  #pragma unroll
  for (int i = 0; i < 2; i++)
    #pragma unroll
    for (int r = 0; r < 8; r++) {
      const int row = smB + 16 * i + r + hi8;
      const float inv = 1.0f / sL[row];
      const int t = i0 + row;
      #pragma unroll
      for (int j = 0; j < 2; j++) {
        const int d = odB + 16 * j + cl;
        AOh[(((size_t)b * 2048 + t) * 16 + h) * 128 + d] =
            (_Float16)(oacc[i][j][r] * inv);
      }
    }
}

// ---------------------------------------------------------------------------
// Kernel 3: out = attn_out @ w_proj^T.
// A tile (f16, contiguous) is DMA'd by the Tensor Data Mover (one 32x16
// sub-tile per wave, TDM inserts the padded LDS stride); B pipelined via regs.
// ---------------------------------------------------------------------------
__global__ __launch_bounds__(256) void proj_kernel(
    const _Float16* __restrict__ AOh, const float* __restrict__ wproj,
    float* __restrict__ out)
{
  __shared__ _Float16 sA[128 * 40];
  __shared__ _Float16 sB[32 * 136];

  const int tid  = threadIdx.x;
  const int lane = tid & 31;
  const int wv   = tid >> 5;
  const int m0   = blockIdx.y * 128;
  const int f0   = blockIdx.x * 128;
  const int mBase = (wv >> 1) * 32;
  const int nBase = (wv & 1) * 64;

  v8f acc[2][4] = {};
  float4 rB[4];
  #pragma unroll
  for (int u = 0; u < 4; u++) {
    const int cc = tid + 256 * u;
    rB[u] = *(const float4*)(wproj + (size_t)(f0 + (cc >> 3)) * 2048 + (cc & 7) * 4);
  }

  for (int k0 = 0; k0 < 2048; k0 += 32) {
    __syncthreads();
#ifdef USE_TDM
    // each wave DMAs its 16 rows x 32 cols of the A tile (async, TENSORcnt)
    tdm_load_tile_f16(
        (unsigned)(uintptr_t)(sA + wv * 16 * 40),
        AOh + (size_t)(m0 + wv * 16) * 2048 + k0,
        /*tile_w=*/32, /*tile_h=*/16,
        /*tensor_w=*/2048, /*tensor_h=*/4096, /*row_stride=*/2048,
        /*pad_interval(16 DW)=*/3, /*pad_amount(4 DW)=*/3);
#else
    for (int cc = tid; cc < 128 * 4; cc += 256) {
      const int r = cc >> 2, c8 = (cc & 3) * 8;
      v8h v = *(const v8h*)(AOh + (size_t)(m0 + r) * 2048 + k0 + c8);
      *(v8h*)(sA + r * 40 + c8) = v;
    }
#endif
    // B regs -> LDS transposed [k][n] (fp32 -> f16)
    #pragma unroll
    for (int u = 0; u < 4; u++) {
      const int cc = tid + 256 * u;
      const int n = cc >> 3, c4 = (cc & 7) * 4;
      sB[(c4 + 0) * 136 + n] = (_Float16)rB[u].x;
      sB[(c4 + 1) * 136 + n] = (_Float16)rB[u].y;
      sB[(c4 + 2) * 136 + n] = (_Float16)rB[u].z;
      sB[(c4 + 3) * 136 + n] = (_Float16)rB[u].w;
    }
    if (k0 + 32 < 2048) {
      #pragma unroll
      for (int u = 0; u < 4; u++) {
        const int cc = tid + 256 * u;
        rB[u] = *(const float4*)(wproj + (size_t)(f0 + (cc >> 3)) * 2048 + k0 + 32 + (cc & 7) * 4);
      }
    }
#ifdef USE_TDM
    __builtin_amdgcn_s_wait_tensorcnt((short)0);
#endif
    __syncthreads();

    v16h af[2], bf[4];
    #pragma unroll
    for (int i = 0; i < 2; i++) af[i] = ldsA_frag(sA, 40, mBase + 16 * i, lane);
    #pragma unroll
    for (int j = 0; j < 4; j++) bf[j] = ldsB_frag(sB, 136, nBase + 16 * j, lane);
    #pragma unroll
    for (int i = 0; i < 2; i++)
      #pragma unroll
      for (int j = 0; j < 4; j++)
        acc[i][j] = wmma_f16(af[i], bf[j], acc[i][j]);
  }

  const int cl = lane & 15, hi8 = (lane >> 4) * 8;
  #pragma unroll
  for (int i = 0; i < 2; i++)
    #pragma unroll
    for (int j = 0; j < 4; j++)
      #pragma unroll
      for (int r = 0; r < 8; r++) {
        const int row = m0 + mBase + 16 * i + r + hi8;
        const int col = f0 + nBase + 16 * j + cl;
        out[(size_t)row * 2048 + col] = acc[i][j][r];
      }
}

// ---------------------------------------------------------------------------
extern "C" void kernel_launch(void* const* d_in, const int* in_sizes, int n_in,
                              void* d_out, int out_size, void* d_ws, size_t ws_size,
                              hipStream_t stream) {
  (void)in_sizes; (void)n_in; (void)out_size; (void)ws_size;
  const float* x     = (const float*)d_in[0];   // [B,T,C] fp32
  const float* wqkv  = (const float*)d_in[1];   // [3C,C]  fp32
  const float* wproj = (const float*)d_in[2];   // [C,C]   fp32
  float* out = (float*)d_out;                   // [B,T,C] fp32

  // f16 workspace: Q [B,H,T,D] | K^T [B,H,D,T] | V [B,H,T,D] | AO [B,T,C]
  const size_t SZ = (size_t)2 * 16 * 2048 * 128;  // 8,388,608 elems
  _Float16* base = (_Float16*)d_ws;
  _Float16* Qh  = base;
  _Float16* KTh = base + SZ;
  _Float16* Vh  = base + 2 * SZ;
  _Float16* AOh = base + 3 * SZ;

  qkv_rope_kernel  <<<dim3(48, 32), 256, 0, stream>>>(x, wqkv, Qh, KTh, Vh);
  flash_attn_kernel<<<dim3(32, 32), 256, 0, stream>>>(Qh, KTh, Vh, AOh);
  proj_kernel      <<<dim3(16, 32), 256, 0, stream>>>(AOh, wproj, out);
}
